// SphericalConv2d_28475633172938
// MI455X (gfx1250) — matrix operand order repaired and checked by reading, not compiled, not run on hardware
//
#include <hip/hip_runtime.h>
#include <stddef.h>
#include <stdint.h>

// ---------------------------------------------------------------------------
// SFNO-style spherical conv for MI455X (gfx1250, wave32, WMMA).
// Truncation-aware: only m<64 freqs and 128 of 256 l-modes ever matter.
// Big stages: bf16 WMMA GEMMs (f32 accumulate). B operand tiles are DMA'd
// into LDS with GLOBAL_LOAD_ASYNC_TO_LDS_B64 (ASYNCcnt) when available,
// overlapping with vectorized (b128) A-tile staging + f32->bf16 convert.
// ---------------------------------------------------------------------------

typedef __bf16 bf16_t;
typedef bf16_t v16bf __attribute__((ext_vector_type(16)));
typedef float  v8f   __attribute__((ext_vector_type(8)));

#define PI_D 3.14159265358979323846

__device__ __forceinline__ bf16_t f2bf(float f) {
    union { float f; unsigned u; } v; v.f = f;
    unsigned r = (v.u + 0x7fffu + ((v.u >> 16) & 1u)) >> 16;
    unsigned short h = (unsigned short)r;
    return __builtin_bit_cast(bf16_t, h);
}
__device__ __forceinline__ unsigned pack2bf(float a, float b) {
    union { float f; unsigned u; } x, y; x.f = a; y.f = b;
    unsigned ra = (x.u + 0x7fffu + ((x.u >> 16) & 1u)) >> 16;
    unsigned rb = (y.u + 0x7fffu + ((y.u >> 16) & 1u)) >> 16;
    return (ra & 0xffffu) | (rb << 16);
}

// ---- CDNA5 async global->LDS DMA (guarded; falls back to sync copy) -------
// clang signature (from diagnostic): (V2i AS1*, V2i AS3*, imm int, imm int)
#if __has_builtin(__builtin_amdgcn_global_load_async_to_lds_b64)
#define HAVE_ASYNC_LDS 1
typedef int gv2i __attribute__((vector_size(8)));                 // GNU V2i
typedef __attribute__((address_space(1))) gv2i as1_v2i;
typedef __attribute__((address_space(3))) gv2i as3_v2i;
__device__ __forceinline__ as1_v2i* to_as1(const void* p) {
    return (as1_v2i*)(unsigned long long)(uintptr_t)p;
}
__device__ __forceinline__ as3_v2i* to_as3(void* p) {
    return (as3_v2i*)(unsigned)(uintptr_t)p;   // low 32 bits of flat = LDS offset
}
#define ASYNC_LDS_B64(g, l) __builtin_amdgcn_global_load_async_to_lds_b64(to_as1(g), to_as3(l), 0, 0)
#if __has_builtin(__builtin_amdgcn_s_wait_asynccnt)
#define WAIT_ASYNC() __builtin_amdgcn_s_wait_asynccnt(0)
#else
#define WAIT_ASYNC() asm volatile("s_wait_asynccnt 0x0" ::: "memory")
#endif
#endif

// Problem constants: B=8, Cin=Cout=32, H(nlat)=256, W(nlon)=512, M1=M2=64.
// mmax used = 64 freqs, l' = 128 kept modes (l<64 and l>=192 -> l'=l-128).

// ---------------- workspace layout (bytes) ----------------
static constexpr size_t OFF_W   = 0;                                     // 256 f32 CC weights
static constexpr size_t OFF_DFT = 1024;                                  // bf16 [128 t][512 w]  (col-major B)
static constexpr size_t OFF_IRF = OFF_DFT + (size_t)128*512*2;           // bf16 [512 w][128 t]  (col-major B)
static constexpr size_t OFF_LF  = OFF_IRF + (size_t)512*128*2;           // bf16 [64 m][128 l'][256 k]
static constexpr size_t OFF_LI  = OFF_LF  + (size_t)64*128*256*2;        // bf16 [64 m][256 k][128 l']
static constexpr size_t OFF_C1  = OFF_LI  + (size_t)64*256*128*2;        // f32  [65536 (bc,k)][128 t]
static constexpr size_t OFF_XFT = OFF_C1  + (size_t)65536*128*4;         // f32  [64 m][512 (bc,part)][128 l']
static constexpr size_t OFF_OFT = OFF_XFT + (size_t)64*512*128*4;        // f32  [64 m][512 (bo,part)][128 l']
static constexpr size_t OFF_XK  = OFF_OFT + (size_t)64*512*128*4;        // f32  [64 m][512 (bo,part)][256 k]
// total ~109.3 MB

// ---------------- setup kernels (f64 math, tiny) ----------------

// Clenshaw-Curtis weights on 256 equiangular nodes (N=255, all b=2).
__global__ void cc_weights_kernel(float* __restrict__ wout) {
    int k = threadIdx.x;                      // 0..255
    double tj = PI_D * (double)k / 255.0;
    double s = 0.0;
    for (int j = 1; j <= 127; ++j)
        s += (2.0 / (4.0 * (double)j * j - 1.0)) * cos(2.0 * j * tj);
    double w = (2.0 / 255.0) * (1.0 - s);
    if (k == 0 || k == 255) w *= 0.5;
    wout[k] = (float)w;
}

// Forward-DFT (scaled 2*pi/512) and irFFT reconstruction matrices, bf16,
// both stored "column-major" ([n][k]) for contiguous async B-tile rows.
__global__ void dft_mats_kernel(bf16_t* __restrict__ dftBc, bf16_t* __restrict__ irfBc) {
    int idx = blockIdx.x * blockDim.x + threadIdx.x;   // 65536 = 128*512
    int t  = idx >> 9;          // 2*m + part  (0..127)
    int w_ = idx & 511;         // longitude sample
    int m  = t >> 1;
    int part = t & 1;
    double th = 2.0 * PI_D * (double)m * (double)w_ / 512.0;
    double cv = part ? -sin(th) : cos(th);
    dftBc[(size_t)t * 512 + w_] = f2bf((float)((2.0 * PI_D / 512.0) * cv)); // [t][w]
    double cm = (m == 0) ? 1.0 : 2.0;   // Hermitian doubling; m=0 imag col -> 0
    irfBc[(size_t)w_ * 128 + t] = f2bf((float)(cm * cv));                   // [w][t]
}

// Orthonormal associated Legendre recurrence (f64), storing only the two
// kept l-bands, bf16, in the column-major B layouts of their GEMMs.
__global__ void legendre_kernel(const float* __restrict__ ccw,
                                bf16_t* __restrict__ legFc,   // [m][l'][k] (fwd, weighted)
                                bf16_t* __restrict__ legIc) { // [m][k][l'] (inv)
    int idx = blockIdx.x * blockDim.x + threadIdx.x;   // 16384 = 64*256
    int m = idx >> 8;           // 0..63
    int k = idx & 255;          // latitude
    double t = PI_D * (double)k / 255.0;
    double cost = cos(t);
    double s2 = 1.0 - cost * cost;
    double sint = sqrt(s2 > 0.0 ? s2 : 0.0);
    double wk = (double)ccw[k];

    auto store = [&](int l, double p) {
        if (l < 64 || l >= 192) {
            int lp = (l < 64) ? l : (l - 128);
            legFc[((size_t)m * 128 + lp) * 256 + k] = f2bf((float)(p * wk));
            legIc[((size_t)m * 256 + k) * 128 + lp] = f2bf((float)p);
        }
    };

    double pmm = sqrt(1.0 / (4.0 * PI_D));
    for (int mm = 1; mm <= m; ++mm)
        pmm = -sqrt((2.0 * mm + 1.0) / (2.0 * mm)) * sint * pmm;

    double p0 = pmm;                                   // l = m
    store(m, p0);
    double p1 = sqrt(2.0 * m + 3.0) * cost * pmm;      // l = m+1
    if (m + 1 < 256) store(m + 1, p1);
    for (int l = m + 2; l < 256; ++l) {
        double dl = (double)l, dm = (double)m;
        double a = sqrt((4.0 * dl * dl - 1.0) / (dl * dl - dm * dm));
        double c = sqrt(((dl - 1.0) * (dl - 1.0) - dm * dm) /
                        (4.0 * (dl - 1.0) * (dl - 1.0) - 1.0));
        double p = a * (cost * p1 - c * p0);
        store(l, p);
        p0 = p1; p1 = p;
    }
}

// ---------------- WMMA GEMM (bf16 operands, f32 accumulate) ----------------
// One 64x64 C tile per 128-thread block (4 waves; wave w -> 16 rows, 64 cols).
// MODE selects addressing for the four pipeline GEMMs:
//   0: DFT      C1[65536,128]  = x[65536,512]        * dftBc'[512,128]
//   1: fwd Leg  xft[m][512,128]= C1-gather[512,256]  * legFc'[m][256,128]
//   2: inv Leg  xk[m][512,256] = oft[m][512,128]     * legIc'[m][128,256]
//   3: irFFT    out[65536,512] = xk-gather[65536,128]* irfBc'[128,512]
// (B matrices stored column-major [n][k]; tiles are contiguous 64B rows.)

union V16 { v16bf v; unsigned u[8]; };

template <int MODE>
__global__ __launch_bounds__(128) void gemm_bf16_wmma(const float* __restrict__ A,
                                                      const bf16_t* __restrict__ B,
                                                      float* __restrict__ C, int Ktot) {
    __shared__ __align__(16) bf16_t As[64 * 34];   // [row][k], 17-dword stride (bank-clean)
    __shared__ __align__(16) bf16_t Bst[64 * 36];  // [col][k], 18-dword stride, 8B-aligned rows

    const int tid  = threadIdx.x;
    const int lane = tid & 31;
    const int wv   = tid >> 5;
    const int zb   = blockIdx.z;                 // batched m for MODE 1/2
    const int Mbase = blockIdx.y * 64;
    const int Nbase = blockIdx.x * 64;

    v8f acc[4];
#pragma unroll
    for (int i = 0; i < 4; ++i) acc[i] = (v8f){0.f,0.f,0.f,0.f,0.f,0.f,0.f,0.f};

    for (int kt = 0; kt < Ktot; kt += 32) {
        // ---- B tile: 64 rows x 32 k of bf16; each row = 64B contiguous ----
#pragma unroll
        for (int j = 0; j < 4; ++j) {
            int i = tid + j * 128;          // 512 x 8-byte chunks
            int n = i >> 3, c = i & 7;      // row (N), chunk (4 bf16)
            size_t gidx;
            if (MODE == 0)      gidx = (size_t)(Nbase + n) * 512 + kt + c * 4;
            else if (MODE == 1) gidx = ((size_t)zb * 128 + Nbase + n) * 256 + kt + c * 4;
            else if (MODE == 2) gidx = ((size_t)zb * 256 + Nbase + n) * 128 + kt + c * 4;
            else                gidx = (size_t)(Nbase + n) * 128 + kt + c * 4;
#ifdef HAVE_ASYNC_LDS
            ASYNC_LDS_B64(B + gidx, &Bst[n * 36 + c * 4]);
#else
            const uint2* g = (const uint2*)(B + gidx);
            uint2 v = *g;
            unsigned* d = (unsigned*)Bst + n * 18 + c * 2;
            d[0] = v.x; d[1] = v.y;
#endif
        }

        // ---- A tile: 64 rows x 32 k, f32 -> bf16 (vectorized per mode) ----
        if (MODE == 0 || MODE == 2) {
            const size_t rowStride = (MODE == 0) ? 512 : 128;
            const float* Abase = (MODE == 0) ? (A + (size_t)Mbase * 512 + kt)
                                             : (A + ((size_t)zb * 512 + Mbase) * 128 + kt);
#pragma unroll
            for (int j = 0; j < 4; ++j) {
                int i = tid + j * 128; int r = i >> 3, c = i & 7;
                float4 v = *(const float4*)(Abase + (size_t)r * rowStride + c * 4);
                unsigned* d = (unsigned*)As + r * 17 + c * 2;
                d[0] = pack2bf(v.x, v.y); d[1] = pack2bf(v.z, v.w);
            }
        } else if (MODE == 1) {
            // rows (bc,part): re/im contiguous pairs; k has stride 128
#pragma unroll
            for (int j = 0; j < 8; ++j) {
                int i = tid + j * 128; int bc = i >> 5, kk = i & 31;
                float2 v = *(const float2*)(A + (((size_t)(Mbase >> 1) + bc) * 256
                                                 + (size_t)(kt + kk)) * 128 + 2 * zb);
                As[(bc * 2) * 34 + kk]     = f2bf(v.x);
                As[(bc * 2 + 1) * 34 + kk] = f2bf(v.y);
            }
        } else { // MODE == 3: contiguous along latitude (rows)
#pragma unroll
            for (int j = 0; j < 4; ++j) {
                int i = tid + j * 128; int kk = i >> 4, rc = i & 15;
                int K = kt + kk; int rg = Mbase + rc * 4;
                float4 v = *(const float4*)(A + ((size_t)(K >> 1) * 512 + (rg >> 8) * 2
                                                 + (K & 1)) * 256 + (rg & 255));
                As[(rc * 4 + 0) * 34 + kk] = f2bf(v.x);
                As[(rc * 4 + 1) * 34 + kk] = f2bf(v.y);
                As[(rc * 4 + 2) * 34 + kk] = f2bf(v.z);
                As[(rc * 4 + 3) * 34 + kk] = f2bf(v.w);
            }
        }
#ifdef HAVE_ASYNC_LDS
        WAIT_ASYNC();
#endif
        __syncthreads();

        // ---- fragments + 4 WMMA (ISA 16-bit A/B VGPR layouts) ----
        V16 af;
        {
            const unsigned* Arow32 = ((const unsigned*)As) + (wv * 16 + (lane & 15)) * 17;
            const int abase = (lane >> 4) * 4;
#pragma unroll
            for (int v2 = 0; v2 < 8; ++v2)
                af.u[v2] = Arow32[abase + ((v2 < 4) ? v2 : v2 + 4)];
        }
#pragma unroll
        for (int nt = 0; nt < 4; ++nt) {
            const unsigned* Brow32 = ((const unsigned*)Bst) + (nt * 16 + (lane & 15)) * 18;
            const int bbase = (lane >> 4) * 4;
            V16 bfr;
#pragma unroll
            for (int v2 = 0; v2 < 8; ++v2) bfr.u[v2] = Brow32[bbase + ((v2 < 4) ? v2 : v2 + 4)];
            acc[nt] = __builtin_amdgcn_wmma_f32_16x16x32_bf16(
                false, af.v, false, bfr.v, (short)0, acc[nt], false, false);
        }
        __syncthreads();
    }

    // store D (ISA 32-bit C/D 16x16 layout): lane&15 = col, v + 8*(lane>>4) = row
    const int rowBase = Mbase + wv * 16 + ((lane < 16) ? 0 : 8);
    const int colBase = Nbase + (lane & 15);
#pragma unroll
    for (int nt = 0; nt < 4; ++nt) {
        int col = colBase + nt * 16;
#pragma unroll
        for (int v2 = 0; v2 < 8; ++v2) {
            int row = rowBase + v2;
            float val = acc[nt][v2];
            if (MODE == 0)      C[(size_t)row * 128 + col] = val;
            else if (MODE == 1) C[((size_t)zb * 512 + row) * 128 + col] = val;
            else if (MODE == 2) C[((size_t)zb * 512 + row) * 256 + col] = val;
            else                C[(size_t)row * 512 + col] = val;
        }
    }
}

// ---------------- complex channel mixing (per mode (l',m): [8x32]x[32x32]) ----
// 1.5% of total FLOPs; VALU f32 complex FMA, one block per mode.
__global__ __launch_bounds__(256) void mix_kernel(const float* __restrict__ xft,
                                                  const float* __restrict__ w1,
                                                  const float* __restrict__ w2,
                                                  float* __restrict__ oft) {
    const int lp = blockIdx.x;   // 0..127 kept l-mode
    const int m  = blockIdx.y;   // 0..63 freq
    __shared__ float xs[512];    // [(b*32+i)*2 + part]
    int tid = threadIdx.x;
    xs[tid]       = xft[((size_t)m * 512 + tid) * 128 + lp];
    xs[tid + 256] = xft[((size_t)m * 512 + tid + 256) * 128 + lp];
    __syncthreads();

    const int b = tid >> 5;      // 0..7
    const int o = tid & 31;      // 0..31
    const float* wsel = (lp < 64) ? w1 : w2;
    const int lx = (lp < 64) ? lp : (lp - 64);

    float ar = 0.f, ai = 0.f;
    for (int i = 0; i < 32; ++i) {
        float xr = xs[(b * 32 + i) * 2];
        float xi = xs[(b * 32 + i) * 2 + 1];
        size_t wa = ((((size_t)i * 32 + o) * 64 + lx) * 64 + m) * 2;
        float wr = wsel[wa], wi = wsel[wa + 1];
        ar += xr * wr - xi * wi;
        ai += xr * wi + xi * wr;
    }
    size_t rowO = (size_t)(b * 32 + o) * 2;
    oft[((size_t)m * 512 + rowO) * 128 + lp]     = ar;
    oft[((size_t)m * 512 + rowO + 1) * 128 + lp] = ai;
}

// ---------------------------------------------------------------------------

extern "C" void kernel_launch(void* const* d_in, const int* in_sizes, int n_in,
                              void* d_out, int out_size, void* d_ws, size_t ws_size,
                              hipStream_t stream) {
    (void)in_sizes; (void)n_in; (void)out_size; (void)ws_size;
    const float* x  = (const float*)d_in[0];   // [8,32,256,512]
    const float* w1 = (const float*)d_in[1];   // [32,32,64,64,2]
    const float* w2 = (const float*)d_in[2];
    float* out = (float*)d_out;                // [8,32,256,512]

    char* ws = (char*)d_ws;
    float*  wsW   = (float*)(ws + OFF_W);
    bf16_t* dftBc = (bf16_t*)(ws + OFF_DFT);
    bf16_t* irfBc = (bf16_t*)(ws + OFF_IRF);
    bf16_t* legFc = (bf16_t*)(ws + OFF_LF);
    bf16_t* legIc = (bf16_t*)(ws + OFF_LI);
    float*  C1    = (float*)(ws + OFF_C1);
    float*  xft   = (float*)(ws + OFF_XFT);
    float*  oft   = (float*)(ws + OFF_OFT);
    float*  xk    = (float*)(ws + OFF_XK);

    // setup (operand matrices, f64 internally)
    cc_weights_kernel<<<1, 256, 0, stream>>>(wsW);
    dft_mats_kernel<<<256, 256, 0, stream>>>(dftBc, irfBc);
    legendre_kernel<<<128, 128, 0, stream>>>(wsW, legFc, legIc);

    // 1) truncated real DFT over longitude:  C1 = x * dftB
    gemm_bf16_wmma<0><<<dim3(2, 1024, 1), 128, 0, stream>>>(x, dftBc, C1, 512);
    // 2) forward Legendre (per m):           xft = C1-gather * legF[m]
    gemm_bf16_wmma<1><<<dim3(2, 8, 64), 128, 0, stream>>>(C1, legFc, xft, 256);
    // 3) complex channel mixing on kept modes
    mix_kernel<<<dim3(128, 64, 1), 256, 0, stream>>>(xft, w1, w2, oft);
    // 4) inverse Legendre (per m):           xk = oft * legI[m]
    gemm_bf16_wmma<2><<<dim3(4, 8, 64), 128, 0, stream>>>(oft, legIc, xk, 128);
    // 5) irFFT over longitude:               out = xk-gather * irfB
    gemm_bf16_wmma<3><<<dim3(8, 1024, 1), 128, 0, stream>>>(xk, irfBc, out, 128);
}
